// MultiHeadAttention_12945031430723
// MI455X (gfx1250) — matrix operand order repaired and checked by reading, not compiled, not run on hardware
//
#include <hip/hip_runtime.h>
#include <hip/hip_bf16.h>

typedef __bf16 bf16;
typedef bf16  v16bf __attribute__((ext_vector_type(16)));
typedef float v8f   __attribute__((ext_vector_type(8)));

#define SEQ_Q 2048
#define SEQ_K 2048
#define BATCH 4
#define HEADS 16
#define HD    64
#define DIM   1024
#define SSTR  2052            // padded LDS row stride (floats) -> bank-conflict free
#define QK_SCALE 0.125f       // 1/sqrt(64)

static __device__ __forceinline__ v8f wmma_bf16(v16bf a, v16bf b, v8f c) {
    // (neg_a, A, neg_b, B, c_mod, C, reuse_a, reuse_b)
    return __builtin_amdgcn_wmma_f32_16x16x32_bf16(false, a, false, b, (short)0, c, false, false);
}

// One workgroup = one (batch, head, 16-row q-tile). 8 waves.
__global__ __launch_bounds__(256)
void mha_attn_kernel(const float* __restrict__ Q, const float* __restrict__ K,
                     const float* __restrict__ V, const unsigned char* __restrict__ mask,
                     float* __restrict__ attn_out, bf16* __restrict__ xbuf)
{
    extern __shared__ float smem[];
    float* scores = smem;                 // 16 * SSTR floats (~128 KB of the 320 KB WGP LDS)
    float* red    = smem + 16 * SSTR;     // 16 * 16
    float* rowmax = red + 256;            // 16
    float* rowsum = rowmax + 16;          // 16

    const int tid  = threadIdx.x;
    const int lane = tid & 31;
    const int wave = tid >> 5;
    const int hl   = lane & 15;           // lane within half-wave
    const int hi   = lane >> 4;           // which half-wave
    const int h    = blockIdx.x & 15;
    const int b    = blockIdx.x >> 4;
    const int q0   = blockIdx.y * 16;

    // ---------------- Phase 1: S = scale * Q K^T, masked, into LDS ----------------
    // A-frag (16x32 bf16): lane<16 holds row q0+hl, k = 0..15; lane>=16: k = 16..31
    const float* qp = Q + ((size_t)(q0 + hl) * BATCH + b) * DIM + h * HD;
    v16bf A0, A1;
    #pragma unroll
    for (int j = 0; j < 8; ++j) {
        const int d0 = (hi ? 16 : 0) + j;
        A0[j]     = (bf16)qp[d0];
        A0[j + 8] = (bf16)qp[8 + d0];
        A1[j]     = (bf16)qp[32 + d0];
        A1[j + 8] = (bf16)qp[40 + d0];
    }

    for (int kt = wave; kt < SEQ_K / 16; kt += 8) {
        const int k0 = kt * 16;
        // B-frag (32x16): lane n holds column n = K-row (k0+hl); contiguous d reads
        const float* kp = K + ((size_t)(k0 + hl) * BATCH + b) * DIM + h * HD + hi * 16;
        v16bf B0, B1;
        #pragma unroll
        for (int j = 0; j < 16; ++j) {
            B0[j] = (bf16)kp[j];
            B1[j] = (bf16)kp[32 + j];
        }
        v8f acc = {0.f,0.f,0.f,0.f,0.f,0.f,0.f,0.f};
        acc = wmma_bf16(A0, B0, acc);
        acc = wmma_bf16(A1, B1, acc);
        #pragma unroll
        for (int r = 0; r < 8; ++r) {
            const int m = r + hi * 8;               // C/D layout: m = r (+8 for hi half)
            float s = acc[r] * QK_SCALE;
            // mask is streamed once, >L2 -> non-temporal
            const unsigned char mk = __builtin_nontemporal_load(
                &mask[((size_t)(q0 + m) * SEQ_K + (k0 + hl)) * BATCH + b]);
            scores[m * SSTR + k0 + hl] = mk ? s : -1e30f;
        }
    }
    __syncthreads();

    // ---------------- Phase 2: exact softmax per row, write attn ----------------
    {
        const int row = tid >> 4;                   // 16 rows
        const int seg = tid & 15;                   // 16 segments of 128 cols
        float* sr = scores + row * SSTR + seg * 128;
        float mx = -3.0e38f;
        #pragma unroll 8
        for (int c = 0; c < 128; ++c) mx = fmaxf(mx, sr[c]);
        red[row * 16 + seg] = mx;
        __syncthreads();
        if (seg == 0) {
            float m2 = red[row * 16];
            #pragma unroll
            for (int i = 1; i < 16; ++i) m2 = fmaxf(m2, red[row * 16 + i]);
            rowmax[row] = m2;
        }
        __syncthreads();
        const float rm = rowmax[row];
        float sum = 0.f;
        #pragma unroll 8
        for (int c = 0; c < 128; ++c) {
            const float e = __expf(sr[c] - rm);
            sr[c] = e;
            sum += e;
        }
        red[row * 16 + seg] = sum;
        __syncthreads();
        if (seg == 0) {
            float s2 = 0.f;
            #pragma unroll
            for (int i = 0; i < 16; ++i) s2 += red[row * 16 + i];
            rowsum[row] = s2;
        }
        __syncthreads();
        const float inv = 1.0f / rowsum[row];
        // attn layout: [q, k, b, h] -- 1.07 GB streamed output, never re-read: NT stores
        float* ap = attn_out +
            (((size_t)(q0 + row) * SEQ_K + seg * 128) * BATCH + b) * HEADS + h;
        #pragma unroll 4
        for (int c = 0; c < 128; ++c) {
            const float p = sr[c] * inv;
            sr[c] = p;
            __builtin_nontemporal_store(p, &ap[(size_t)c * BATCH * HEADS]);
        }
    }
    __syncthreads();

    // ---------------- Phase 3: x = P @ V  (wave-split over K, 4 n-tiles) ----------------
    v8f xacc[4];
    #pragma unroll
    for (int nt = 0; nt < 4; ++nt) xacc[nt] = (v8f){0.f,0.f,0.f,0.f,0.f,0.f,0.f,0.f};

    for (int kc = 0; kc < 8; ++kc) {
        const int kb = (wave * 8 + kc) * 32;
        v16bf A;
        #pragma unroll
        for (int j = 0; j < 8; ++j) {
            const int kk = kb + (hi ? 16 : 0) + j;
            A[j]     = (bf16)scores[hl * SSTR + kk];
            A[j + 8] = (bf16)scores[hl * SSTR + kk + 8];
        }
        #pragma unroll
        for (int nt = 0; nt < 4; ++nt) {
            const float* vp = V + ((size_t)(kb + hi * 16) * BATCH + b) * DIM
                                + h * HD + nt * 16 + hl;
            v16bf Bv;
            #pragma unroll
            for (int j = 0; j < 16; ++j) Bv[j] = (bf16)vp[(size_t)j * BATCH * DIM];
            xacc[nt] = wmma_bf16(A, Bv, xacc[nt]);
        }
    }
    __syncthreads();            // all waves done reading `scores`; reuse LDS

    float* part = smem;         // [8 waves][16 m][64 d]
    #pragma unroll
    for (int nt = 0; nt < 4; ++nt)
        #pragma unroll
        for (int r = 0; r < 8; ++r)
            part[(wave * 16 + r + hi * 8) * 64 + nt * 16 + hl] = xacc[nt][r];
    __syncthreads();

    for (int i = tid; i < 16 * 64; i += 256) {
        const int m = i >> 6, d = i & 63;
        float s = 0.f;
        #pragma unroll
        for (int w = 0; w < 8; ++w) s += part[(w * 16 + m) * 64 + d];
        const size_t row = (size_t)(q0 + m) * BATCH + b;
        xbuf[row * DIM + h * HD + d] = (bf16)s;
    }
}

// out[row, col] = sum_k x[row,k] * W[col,k] + bias[col];  rows = 8192, cols = 1024
__global__ __launch_bounds__(256)
void mha_proj_kernel(const bf16* __restrict__ xbuf, const float* __restrict__ W,
                     const float* __restrict__ bias, float* __restrict__ out)
{
    const int lane = threadIdx.x & 31;
    const int wave = threadIdx.x >> 5;
    const int hl   = lane & 15;
    const int hi   = lane >> 4;
    const int r0   = blockIdx.y * 16;
    const int c0   = blockIdx.x * 128 + wave * 16;

    const bf16*  xr = xbuf + (size_t)(r0 + hl) * DIM;   // A row (bf16, contiguous)
    const float* wr = W    + (size_t)(c0 + hl) * DIM;   // B column = W row (contiguous)

    v8f acc = {0.f,0.f,0.f,0.f,0.f,0.f,0.f,0.f};
    for (int kb = 0; kb < DIM; kb += 32) {
        v16bf A, B;
        #pragma unroll
        for (int j = 0; j < 8; ++j) {
            const int kk = kb + (hi ? 16 : 0) + j;
            A[j]     = xr[kk];
            A[j + 8] = xr[kk + 8];
        }
        #pragma unroll
        for (int j = 0; j < 16; ++j)
            B[j] = (bf16)wr[kb + hi * 16 + j];
        acc = wmma_bf16(A, B, acc);
    }
    #pragma unroll
    for (int r = 0; r < 8; ++r) {
        const int row = r0 + r + hi * 8;
        const int col = c0 + hl;
        // final output streamed once -> NT store
        __builtin_nontemporal_store(acc[r] + bias[col], &out[(size_t)row * DIM + col]);
    }
}

extern "C" void kernel_launch(void* const* d_in, const int* in_sizes, int n_in,
                              void* d_out, int out_size, void* d_ws, size_t ws_size,
                              hipStream_t stream)
{
    (void)in_sizes; (void)n_in; (void)out_size; (void)ws_size;
    const float*         Q    = (const float*)d_in[0];
    const float*         K    = (const float*)d_in[1];
    const float*         V    = (const float*)d_in[2];
    const unsigned char* mask = (const unsigned char*)d_in[3];
    const float*         W    = (const float*)d_in[4];
    const float*         bias = (const float*)d_in[5];

    float* out  = (float*)d_out;
    float* attn = out + (size_t)SEQ_Q * BATCH * DIM;    // outputs concatenated: out, attn
    bf16*  xbuf = (bf16*)d_ws;                          // 8192*1024 bf16 = 16 MiB scratch

    const size_t lds = (size_t)(16 * SSTR + 256 + 16 + 16) * sizeof(float);

    mha_attn_kernel<<<dim3(BATCH * HEADS, SEQ_Q / 16), 256, lds, stream>>>(
        Q, K, V, mask, attn, xbuf);
    mha_proj_kernel<<<dim3(DIM / 128, (SEQ_Q * BATCH) / 16), 256, 0, stream>>>(
        xbuf, W, bias, out);
}
